// GATv2_48455821033978
// MI455X (gfx1250) — matrix operand order repaired and checked by reading, not compiled, not run on hardware
//
#include <hip/hip_runtime.h>

typedef __attribute__((ext_vector_type(16))) _Float16 v16h;
typedef __attribute__((ext_vector_type(8)))  _Float16 v8h;
typedef __attribute__((ext_vector_type(8)))  float    v8f;
typedef __attribute__((ext_vector_type(4)))  unsigned u32x4;
typedef __attribute__((ext_vector_type(8)))  int      i32x8;
typedef __attribute__((ext_vector_type(4)))  int      i32x4;

#define NEG_SLOPE 0.2f

// ---------- order-preserving float<->uint key (for atomicMax-based segment_max) ----------
__device__ __forceinline__ unsigned f2key(float f) {
  unsigned u = __float_as_uint(f);
  return (u & 0x80000000u) ? ~u : (u | 0x80000000u);
}
__device__ __forceinline__ float key2f(unsigned k) {
  unsigned u = (k & 0x80000000u) ? (k & 0x7FFFFFFFu) : ~k;
  return __uint_as_float(u);
}

// ---------- f32 -> f16 conversion ----------
__global__ void k_tof16(const float* __restrict__ in, _Float16* __restrict__ out, int n) {
  int i = blockIdx.x * blockDim.x + threadIdx.x;
  if (i < n) out[i] = (_Float16)in[i];
}

// ---------- grid-stride u32 fill (zero / -inf-key init) ----------
__global__ void k_fill_u32(unsigned* __restrict__ p, unsigned v, size_t n) {
  size_t i = (size_t)blockIdx.x * blockDim.x + threadIdx.x;
  size_t stride = (size_t)gridDim.x * blockDim.x;
  for (; i < n; i += stride) p[i] = v;
}

// ---------- WMMA GEMM: Y[M,O] = Xh[M,K] @ Wh[O,K]^T + bias ----------
// Block: up to 4 waves, each computing a 16x64 output strip of the SAME 16-row
// M-tile. The shared 16xK A tile is DMA'd once per block into LDS by the Tensor
// Data Mover (tensor_load_to_lds, TENSORcnt, workgroup barrier), then A fragments
// come from LDS (ds_load_b128) and B fragments from global (L2-resident weights).
// A-fragment layout per ISA 7.12.2 (16-bit A 16x32): lanes 0-15 -> K {0..7,16..23},
// lanes 16-31 -> K {8..15,24..31}. B 32x16: same K layout as A (lanes 0-15 -> K {0..7,16..23}, lanes 16-31 -> K {8..15,24..31}).
template<int K>
__global__ __launch_bounds__(128) void k_gemm_wmma(
    const _Float16* __restrict__ Xh,   // [M,K] row-major f16
    const _Float16* __restrict__ Wh,   // [O,K] row-major f16
    const float*    __restrict__ bias, // [O]
    float*          __restrict__ Y,    // [M,O] row-major f32
    int O)
{
  __shared__ _Float16 lds_a[16 * K];

  const int wave   = threadIdx.x >> 5;
  const int lane   = threadIdx.x & 31;
  const int lh     = lane >> 4;   // half-wave id
  const int lm     = lane & 15;
  const int mbase  = blockIdx.x * 16;
  const int nstrip = blockIdx.y * (blockDim.x >> 5) + wave;  // always valid (host sizes block)
  const int nbase  = nstrip * 64;

  // ---- TDM: DMA the 16xK A tile (row-major, 2B elements) into LDS, wave 0 only ----
  if (wave == 0) {
    unsigned long long ga = (unsigned long long)(uintptr_t)(Xh + (size_t)mbase * K);
    unsigned ldsoff = (unsigned)(uintptr_t)(&lds_a[0]);   // flat-LDS aperture: addr[31:0] = LDS offset
    u32x4 g0;
    g0.x = 1u;                                            // count=1, is_restore=0, gather off
    g0.y = ldsoff;                                        // lds_addr (bytes)
    g0.z = (unsigned)(ga & 0xFFFFFFFFull);                // global_addr[31:0]
    g0.w = (unsigned)((ga >> 32) & 0x01FFFFFFull) | (2u << 30); // global_addr[56:32] | type=2
    i32x8 g1;
    g1[0] = 1 << 16;                                      // workgroup_mask=0, data_size=1 (2B)
    g1[1] = (K & 0xFFFF) << 16;                           // tensor_dim0[15:0]
    g1[2] = (K >> 16) | (16 << 16);                       // tensor_dim0[31:16] | tensor_dim1=16
    g1[3] = (K & 0xFFFF) << 16;                           // tensor_dim1 hi=0 | tile_dim0=K
    g1[4] = 16;                                           // tile_dim1=16, tile_dim2=0
    g1[5] = K;                                            // tensor_dim0_stride[31:0] = K
    g1[6] = 0;                                            // stride0 hi | stride1 lo (unused, 2D)
    g1[7] = 0;
    i32x4 z4 = {0, 0, 0, 0};                              // groups 2/3 unused (2D tensor)
    i32x8 z8 = {0, 0, 0, 0, 0, 0, 0, 0};
    __builtin_amdgcn_tensor_load_to_lds(g0, g1, z4, z4, z8, 0);
    __builtin_amdgcn_s_wait_tensorcnt(0);
  }
  __syncthreads();                                        // publish LDS tile to all waves

  const _Float16* arow  = lds_a + (size_t)lm * K;
  const _Float16* brow0 = Wh + (size_t)(nbase +  0 + lm) * K + 8 * lh;
  const _Float16* brow1 = Wh + (size_t)(nbase + 16 + lm) * K + 8 * lh;
  const _Float16* brow2 = Wh + (size_t)(nbase + 32 + lm) * K + 8 * lh;
  const _Float16* brow3 = Wh + (size_t)(nbase + 48 + lm) * K + 8 * lh;

  v8f acc[4] = {};
  v16h a[2];
  v16h b[2][4];

  auto load_step = [&](int buf, int kk) {
    ((v8h*)&a[buf])[0] = *(const v8h*)(arow + kk +      8 * lh);   // ds_load_b128
    ((v8h*)&a[buf])[1] = *(const v8h*)(arow + kk + 16 + 8 * lh);
    ((v8h*)&b[buf][0])[0] = *(const v8h*)(brow0 + kk);
    ((v8h*)&b[buf][0])[1] = *(const v8h*)(brow0 + kk + 16);
    ((v8h*)&b[buf][1])[0] = *(const v8h*)(brow1 + kk);
    ((v8h*)&b[buf][1])[1] = *(const v8h*)(brow1 + kk + 16);
    ((v8h*)&b[buf][2])[0] = *(const v8h*)(brow2 + kk);
    ((v8h*)&b[buf][2])[1] = *(const v8h*)(brow2 + kk + 16);
    ((v8h*)&b[buf][3])[0] = *(const v8h*)(brow3 + kk);
    ((v8h*)&b[buf][3])[1] = *(const v8h*)(brow3 + kk + 16);
  };

  constexpr int NK = K >> 5;
  load_step(0, 0);
#pragma unroll
  for (int i = 0; i < NK - 1; ++i) {
    const int cur = i & 1;
    load_step(cur ^ 1, (i + 1) * 32);          // prefetch next K-step
#pragma unroll
    for (int t = 0; t < 4; ++t)
      acc[t] = __builtin_amdgcn_wmma_f32_16x16x32_f16(
          false, a[cur], false, b[cur][t], (short)0, acc[t], false, false);
  }
  {
    const int cur = (NK - 1) & 1;              // peeled last step
#pragma unroll
    for (int t = 0; t < 4; ++t)
      acc[t] = __builtin_amdgcn_wmma_f32_16x16x32_f16(
          false, a[cur], false, b[cur][t], (short)0, acc[t], false, false);
  }

#pragma unroll
  for (int t = 0; t < 4; ++t) {
    const int col = nbase + t * 16 + lm;
    const float bv = bias[col];
#pragma unroll
    for (int r = 0; r < 8; ++r) {
      const int row = mbase + r + 8 * lh;      // C/D layout: vgpr r, lanes 16-31 -> M+8
      Y[(size_t)row * O + col] = acc[t][r] + bv;
    }
  }
}

// ---------- E1: per-edge logits + segment max (one wave per edge) ----------
__global__ void k_edge_logit(const float* __restrict__ xl, const float* __restrict__ xr,
                             const int* __restrict__ ei, const float* __restrict__ att,
                             unsigned* __restrict__ mkey, float* __restrict__ logit,
                             int E, int N, int H, int C)
{
  int wid  = (blockIdx.x * blockDim.x + threadIdx.x) >> 5;
  int lane = threadIdx.x & 31;
  int Etot = E + N;
  if (wid >= Etot) return;
  int src = (wid < E) ? ei[wid]     : (wid - E);   // self loops appended
  int dst = (wid < E) ? ei[E + wid] : (wid - E);
  int HC = H * C;
  const float* pl = xl + (size_t)src * HC;
  const float* pr = xr + (size_t)dst * HC;
  for (int h = 0; h < H; ++h) {
    float s = 0.f;
    for (int c = lane; c < C; c += 32) {
      float v = pl[h * C + c] + pr[h * C + c];
      v = (v > 0.f) ? v : v * NEG_SLOPE;           // LeakyReLU
      s += v * att[h * C + c];
    }
    for (int off = 16; off > 0; off >>= 1) s += __shfl_xor(s, off, 32);
    if (lane == 0) {
      logit[(size_t)wid * H + h] = s;
      atomicMax(&mkey[dst * H + h], f2key(s));
    }
  }
}

// ---------- E2: ex = exp(logit - m[dst]); den[dst] += ex ----------
__global__ void k_edge_expden(const int* __restrict__ ei, const unsigned* __restrict__ mkey,
                              float* __restrict__ exbuf, float* __restrict__ den,
                              int E, int N, int H)
{
  int idx = blockIdx.x * blockDim.x + threadIdx.x;
  int Etot = E + N;
  if (idx >= Etot * H) return;
  int e = idx / H, h = idx - e * H;
  int dst = (e < E) ? ei[E + e] : (e - E);
  float m  = key2f(mkey[dst * H + h]);
  float ex = expf(exbuf[idx] - m);
  exbuf[idx] = ex;
  atomicAdd(&den[dst * H + h], ex);
}

// ---------- E3: out[dst] += alpha * xl[src] (one wave per edge) ----------
__global__ void k_edge_scatter(const float* __restrict__ xl, const float* __restrict__ exbuf,
                               const float* __restrict__ den, const int* __restrict__ ei,
                               float* __restrict__ out, int E, int N, int H, int C)
{
  int wid  = (blockIdx.x * blockDim.x + threadIdx.x) >> 5;
  int lane = threadIdx.x & 31;
  int Etot = E + N;
  if (wid >= Etot) return;
  int src = (wid < E) ? ei[wid]     : (wid - E);
  int dst = (wid < E) ? ei[E + wid] : (wid - E);
  int HC = H * C;
  const float* pl = xl + (size_t)src * HC;
  float*       po = out + (size_t)dst * HC;
  for (int h = 0; h < H; ++h) {
    float alpha = exbuf[(size_t)wid * H + h] / den[dst * H + h];
    for (int c = lane; c < C; c += 32)
      atomicAdd(&po[h * C + c], alpha * pl[h * C + c]);
  }
}

// ---------- E4: h = acc + bias, optional ELU ----------
__global__ void k_epilogue(const float* __restrict__ acc, const float* __restrict__ bias,
                           float* __restrict__ hout, int n, int HC, int elu)
{
  int i = blockIdx.x * blockDim.x + threadIdx.x;
  if (i >= n) return;
  float v = acc[i] + bias[i % HC];
  if (elu) v = (v > 0.f) ? v : expm1f(v);
  hout[i] = v;
}

// ---------- column sums for global mean pooling ----------
__global__ void k_colsum(const float* __restrict__ h, float* __restrict__ colsum, int n, int C) {
  int c = threadIdx.x;   // blockDim.x == C
  float s = 0.f;
  for (int r = blockIdx.x; r < n; r += gridDim.x) s += h[(size_t)r * C + c];
  atomicAdd(&colsum[c], s);
}

// ---------- final linear on pooled embedding ----------
__global__ void k_final(const float* __restrict__ colsum, const float* __restrict__ lw,
                        const float* __restrict__ lb, float* __restrict__ out,
                        int n, int C, int dout)
{
  int j = threadIdx.x;
  if (j >= dout) return;
  float inv = 1.f / (float)n;
  float s = 0.f;
  for (int c = 0; c < C; ++c) s += (colsum[c] * inv) * lw[j * C + c];
  out[j] = s + lb[j];
}

extern "C" void kernel_launch(void* const* d_in, const int* in_sizes, int n_in,
                              void* d_out, int out_size, void* d_ws, size_t ws_size,
                              hipStream_t stream) {
  (void)in_sizes; (void)n_in; (void)out_size; (void)ws_size;
  const int N = 20000, E = 200000, DIN = 128, DH = 128, HEADS = 4, DOUT = 10;
  const int Etot = E + N;

  const float* x     = (const float*)d_in[0];
  const int*   ei    = (const int*)d_in[1];
  const float* lin_w = (const float*)d_in[20];
  const float* lin_b = (const float*)d_in[21];

  // workspace carve (256B aligned)
  char* wp = (char*)d_ws;
  auto carve = [&](size_t bytes) { void* p = (void*)wp; wp += (bytes + 255) & ~(size_t)255; return p; };
  _Float16* act16 = (_Float16*)carve((size_t)N * 512 * 2);
  _Float16* w16l  = (_Float16*)carve((size_t)512 * 512 * 2);
  _Float16* w16r  = (_Float16*)carve((size_t)512 * 512 * 2);
  float*    xl    = (float*)   carve((size_t)N * 512 * 4);
  float*    xrout = (float*)   carve((size_t)N * 512 * 4);   // xr, later reused as out-accum
  float*    hbuf  = (float*)   carve((size_t)N * 512 * 4);
  float*    exbuf = (float*)   carve((size_t)Etot * HEADS * 4);
  unsigned* mkey  = (unsigned*)carve((size_t)N * HEADS * 4);
  float*    den   = (float*)   carve((size_t)N * HEADS * 4);
  float*    colsum= (float*)   carve(512);

  const float* input = x;
  for (int layer = 0; layer < 3; ++layer) {
    int base = 2 + layer * 6;
    const float* Wl  = (const float*)d_in[base + 0];
    const float* bl  = (const float*)d_in[base + 1];
    const float* Wr  = (const float*)d_in[base + 2];
    const float* br  = (const float*)d_in[base + 3];
    const float* att = (const float*)d_in[base + 4];
    const float* bs  = (const float*)d_in[base + 5];
    const int Kin = (layer == 0) ? DIN : DH * HEADS;
    const int H   = (layer == 2) ? 1 : HEADS;
    const int C   = DH;
    const int O   = H * C;
    const int elu = (layer < 2) ? 1 : 0;

    // f32 -> f16 for WMMA operands
    { int n = N * Kin; k_tof16<<<(n + 255) / 256, 256, 0, stream>>>(input, act16, n); }
    { int n = O * Kin;
      k_tof16<<<(n + 255) / 256, 256, 0, stream>>>(Wl, w16l, n);
      k_tof16<<<(n + 255) / 256, 256, 0, stream>>>(Wr, w16r, n); }

    // xl = act @ Wl^T + bl ; xr = act @ Wr^T + br
    const int strips = O / 64;
    const int wpb    = (strips >= 4) ? 4 : strips;     // waves per block, all active
    dim3 gg(N / 16, strips / wpb);
    dim3 bb(32 * wpb);
    if (Kin == 128) {
      k_gemm_wmma<128><<<gg, bb, 0, stream>>>(act16, w16l, bl, xl,    O);
      k_gemm_wmma<128><<<gg, bb, 0, stream>>>(act16, w16r, br, xrout, O);
    } else {
      k_gemm_wmma<512><<<gg, bb, 0, stream>>>(act16, w16l, bl, xl,    O);
      k_gemm_wmma<512><<<gg, bb, 0, stream>>>(act16, w16r, br, xrout, O);
    }

    // segment softmax over dst + scatter
    k_fill_u32<<<1024, 256, 0, stream>>>(mkey, 0x007FFFFFu /* key(-inf) */, (size_t)N * H);
    k_fill_u32<<<1024, 256, 0, stream>>>((unsigned*)den, 0u, (size_t)N * H);

    int wblocks = (Etot * 32 + 255) / 256;                 // one wave per edge
    k_edge_logit<<<wblocks, 256, 0, stream>>>(xl, xrout, ei, att, mkey, exbuf, E, N, H, C);
    { int n = Etot * H;
      k_edge_expden<<<(n + 255) / 256, 256, 0, stream>>>(ei, mkey, exbuf, den, E, N, H); }

    k_fill_u32<<<2048, 256, 0, stream>>>((unsigned*)xrout, 0u, (size_t)N * O); // zero out-accum
    k_edge_scatter<<<wblocks, 256, 0, stream>>>(xl, exbuf, den, ei, xrout, E, N, H, C);

    { int n = N * O;
      k_epilogue<<<(n + 255) / 256, 256, 0, stream>>>(xrout, bs, hbuf, n, O, elu); }
    input = hbuf;
  }

  // global mean pooling + final linear -> [1, DOUT]
  k_fill_u32<<<1, 128, 0, stream>>>((unsigned*)colsum, 0u, (size_t)DH);
  k_colsum<<<512, DH, 0, stream>>>(hbuf, colsum, N, DH);
  k_final<<<1, 32, 0, stream>>>(colsum, lin_w, lin_b, (float*)d_out, N, DH, DOUT);
}
